// AddTaskModel_27427661152399
// MI455X (gfx1250) — compile-verified, matrix operand order checked by reading, not run on hardware
//
#include <hip/hip_runtime.h>

// CDNA5 / gfx1250, wave32. Adaptive-spiking RNN scan fused into one persistent
// per-batch-chunk kernel using v_wmma_f32_16x16x32_f16.
// v4: - sigmoid via v_rcp_f32 (hardware reciprocal) instead of precise IEEE
//       division -> removes v_div_scale/v_div_fmas chains from the critical path
//     - time loop unrolled x2 so ping-pong buffer selects constant-fold
//     - gate partials from previous state before the dense barrier (12+8 WMMA split)
//     - x_t prefetched one step ahead; 2 barriers/step; split accumulator chains

typedef __attribute__((ext_vector_type(16))) _Float16 v16h;
typedef __attribute__((ext_vector_type(8)))  _Float16 v8h;
typedef __attribute__((ext_vector_type(8)))  float    v8f;

#define HDIM 128
#define DDIM 2
#define PITCH 136   // halfwords per staged A-row: 272 bytes, 16B-aligned rows, conflict-free

// A-operand (16x32 f16) fetch from LDS staging buffer.
// ISA layout: lane l holds M = l&15; half = l>>4 selects K {0..7,16..23} vs {8..15,24..31}.
// Both K runs are 8 contiguous halfwords -> two 16-byte LDS loads.
__device__ __forceinline__ v16h load_a(const _Float16* sA, int m, int half, int ks) {
  const _Float16* p = sA + m * PITCH + ks * 32 + half * 8;
  v8h lo = *(const v8h*)(p);
  v8h hi = *(const v8h*)(p + 16);
  v16h a;
#pragma unroll
  for (int e = 0; e < 8; ++e) { a[e] = lo[e]; a[8 + e] = hi[e]; }
  return a;
}

// B-operand (32x16 f16) built from one f32 weight row (row = output neuron n).
// Symmetric layout: lane l holds N = l&15; half selects K sub-ranges as for A.
__device__ __forceinline__ v16h load_b(const float* __restrict__ Wrow, int ks, int half) {
  v16h b;
  const int kb = ks * 32 + half * 8;
#pragma unroll
  for (int e = 0; e < 8; ++e) b[e]     = (_Float16)Wrow[kb + e];
#pragma unroll
  for (int e = 0; e < 8; ++e) b[8 + e] = (_Float16)Wrow[kb + 16 + e];
  return b;
}

// sigmoid with hardware reciprocal (v_rcp_f32): 1 trans op instead of IEEE divide.
__device__ __forceinline__ float fast_sigmoid(float v) {
  return __builtin_amdgcn_rcpf(1.0f + __expf(-v));
}

#define WMMA(a, b, c) __builtin_amdgcn_wmma_f32_16x16x32_f16(false, (a), false, (b), (short)0, (c), false, false)

__global__ __launch_bounds__(256)
void srnn_scan_kernel(const float* __restrict__ x,      // [S,B,D]
                      const float* __restrict__ yv,     // [B]
                      const float* __restrict__ h0_mem, // [B,H]
                      const float* __restrict__ h0_spk,
                      const float* __restrict__ h0_b,
                      const float* __restrict__ W1x,    // [H, D+H]
                      const float* __restrict__ b1x,    // [H]
                      const float* __restrict__ WtauM,  // [H, 2H]
                      const float* __restrict__ btauM,
                      const float* __restrict__ WtauA,  // [H, 2H]
                      const float* __restrict__ btauA,
                      const float* __restrict__ Wlin,   // [1,H]
                      const float* __restrict__ blin,   // [1]
                      float* __restrict__ sqerr,        // [B] per-row (out-y)^2
                      int S, int Bsz) {
  // Ping-pong staged state + x (read [t&1], write [1-(t&1)]); dense is intra-step only.
  __shared__ __align__(16) _Float16 sSpk[2][16 * PITCH];
  __shared__ __align__(16) _Float16 sMem[2][16 * PITCH];
  __shared__ __align__(16) _Float16 sBb [2][16 * PITCH];
  __shared__ __align__(16) _Float16 sDen[16 * PITCH];
  __shared__ float sX[2][16 * DDIM];
  __shared__ float sM32[16 * HDIM];

  const int tid  = threadIdx.x;
  const int wave = tid >> 5;        // 0..7 -> owns H columns [wave*16, wave*16+16)
  const int lane = tid & 31;
  const int half = lane >> 4;
  const int lq   = lane & 15;
  const int ncol = wave * 16 + lq;  // global hidden index for B/C layouts
  const int row0 = blockIdx.x * 16; // batch rows [row0, row0+16)

  // ---- static per-lane loads: weight B-tiles (kept in VGPRs for all steps) ----
  v16h Bs[4], BM1[4], BM2[4], BA1[4], BA2[4];
#pragma unroll
  for (int ks = 0; ks < 4; ++ks) {
    Bs [ks] = load_b(W1x   + ncol * (DDIM + HDIM) + DDIM, ks, half);
    BM1[ks] = load_b(WtauM + ncol * (2 * HDIM),           ks, half);
    BM2[ks] = load_b(WtauM + ncol * (2 * HDIM) + HDIM,    ks, half);
    BA1[ks] = load_b(WtauA + ncol * (2 * HDIM),           ks, half);
    BA2[ks] = load_b(WtauA + ncol * (2 * HDIM) + HDIM,    ks, half);
  }
  const float wx0 = W1x[ncol * (DDIM + HDIM) + 0];
  const float wx1 = W1x[ncol * (DDIM + HDIM) + 1];
  const float bd  = b1x[ncol];
  const float bM  = btauM[ncol];
  const float bA  = btauA[ncol];

  // ---- f32 state in registers, C/D layout: vgpr r, lane -> row m = r + 8*half, col ncol ----
  v8f mem, spk, bb;
#pragma unroll
  for (int r = 0; r < 8; ++r) {
    const int m  = r + half * 8;
    const int gi = (row0 + m) * HDIM + ncol;
    mem[r] = h0_mem[gi];
    spk[r] = h0_spk[gi];
    bb[r]  = h0_b[gi];
  }
  // stage f16 state into buffer 0 (read at t=0), and x_0 into sX[0]
#pragma unroll
  for (int r = 0; r < 8; ++r) {
    const int m = r + half * 8;
    sSpk[0][m * PITCH + ncol] = (_Float16)spk[r];
    sMem[0][m * PITCH + ncol] = (_Float16)mem[r];
    sBb [0][m * PITCH + ncol] = (_Float16)bb[r];
  }
  if (tid < 16 * DDIM) sX[0][tid] = x[row0 * DDIM + tid];

#pragma unroll 2
  for (int t = 0; t < S; ++t) {
    const int rp = t & 1;       // read buffer
    const int wp = rp ^ 1;      // write buffer (last read at step t-1 -> safe after barriers)
    const _Float16* sSpkR = sSpk[rp];
    const _Float16* sMemR = sMem[rp];
    const _Float16* sBbR  = sBb [rp];
    _Float16* sSpkW = sSpk[wp];
    _Float16* sMemW = sMem[wp];
    _Float16* sBbW  = sBb [wp];

    __syncthreads();  // barrier1: previous-step state + prefetched x visible

    // ---- phase (a): 12 independent WMMAs on previous-step state ----
    //   dense  = b1x + x@Wx^T + spk@Ws^T             (chains accD0/accD1, depth 2)
    //   gateM  = btauM + mem@WM2^T  (partial)         (chains accM0/accM1, depth 2)
    //   gateA  = btauA + bb @WA2^T  (partial)         (chains accA0/accA1, depth 2)
    v16h aS[4], aM[4], aB[4];
#pragma unroll
    for (int ks = 0; ks < 4; ++ks) aS[ks] = load_a(sSpkR, lq, half, ks);
#pragma unroll
    for (int ks = 0; ks < 4; ++ks) aM[ks] = load_a(sMemR, lq, half, ks);
#pragma unroll
    for (int ks = 0; ks < 4; ++ks) aB[ks] = load_a(sBbR,  lq, half, ks);

    v8f accD0, accD1, accM0, accM1, accA0, accA1;
#pragma unroll
    for (int r = 0; r < 8; ++r) {
      const int m = r + half * 8;
      accD0[r] = bd + wx0 * sX[rp][m * DDIM + 0] + wx1 * sX[rp][m * DDIM + 1];
      accD1[r] = 0.0f;
      accM0[r] = bM; accM1[r] = 0.0f;
      accA0[r] = bA; accA1[r] = 0.0f;
    }

    accD0 = WMMA(aS[0], Bs[0],  accD0);
    accD1 = WMMA(aS[1], Bs[1],  accD1);
    accM0 = WMMA(aM[0], BM2[0], accM0);
    accM1 = WMMA(aM[1], BM2[1], accM1);
    accA0 = WMMA(aB[0], BA2[0], accA0);
    accA1 = WMMA(aB[1], BA2[1], accA1);
    accD0 = WMMA(aS[2], Bs[2],  accD0);
    accD1 = WMMA(aS[3], Bs[3],  accD1);
    accM0 = WMMA(aM[2], BM2[2], accM0);
    accM1 = WMMA(aM[3], BM2[3], accM1);
    accA0 = WMMA(aB[2], BA2[2], accA0);
    accA1 = WMMA(aB[3], BA2[3], accA1);

    v8f accD;
#pragma unroll
    for (int r = 0; r < 8; ++r) accD[r] = accD0[r] + accD1[r];
#pragma unroll
    for (int r = 0; r < 8; ++r) {
      const int m = r + half * 8;
      sDen[m * PITCH + ncol] = (_Float16)accD[r];
    }
    __syncthreads();  // barrier2: dense staged

    // prefetch x_{t+1} under the gate WMMAs (buffer last read in step t-1)
    if (t + 1 < S && tid < 16 * DDIM)
      sX[wp][tid] = x[(size_t)(t + 1) * Bsz * DDIM + row0 * DDIM + tid];

    // ---- phase (b): 8 WMMAs on dense (4 chains, depth 2) ----
    v16h aD[4];
#pragma unroll
    for (int ks = 0; ks < 4; ++ks) aD[ks] = load_a(sDen, lq, half, ks);

    accM0 = WMMA(aD[0], BM1[0], accM0);
    accM1 = WMMA(aD[1], BM1[1], accM1);
    accA0 = WMMA(aD[0], BA1[0], accA0);
    accA1 = WMMA(aD[1], BA1[1], accA1);
    accM0 = WMMA(aD[2], BM1[2], accM0);
    accM1 = WMMA(aD[3], BM1[3], accM1);
    accA0 = WMMA(aD[2], BA1[2], accA0);
    accA1 = WMMA(aD[3], BA1[3], accA1);

    // ---- phase (c): elementwise update (registers) + stage next-step f16 state ----
    // Writes target the ping-pong partner; last reads of it were step t-1, ordered
    // by this step's barriers -> no extra barrier needed here.
#pragma unroll
    for (int r = 0; r < 8; ++r) {
      const float gM  = accM0[r] + accM1[r];
      const float gA  = accA0[r] + accA1[r];
      const float tM  = fast_sigmoid(gM);
      const float tA  = fast_sigmoid(gA);
      const float nb  = tA * bb[r] + (1.0f - tA) * spk[r];
      const float Bth = 0.01f + 1.8f * nb;
      const float nm  = mem[r] * tM + (1.0f - tM) * accD[r] - Bth * spk[r];
      const float ns  = (nm - Bth) > 0.0f ? 1.0f : 0.0f;
      bb[r] = nb; mem[r] = nm; spk[r] = ns;
      const int m = r + half * 8;
      sSpkW[m * PITCH + ncol] = (_Float16)ns;
      sMemW[m * PITCH + ncol] = (_Float16)nm;
      sBbW [m * PITCH + ncol] = (_Float16)nb;
    }
  }

  // ---- readout: out[row] = mem[row]@Wlin^T + blin ; emit (out-y)^2 per row ----
#pragma unroll
  for (int r = 0; r < 8; ++r) {
    const int m = r + half * 8;
    sM32[m * HDIM + ncol] = mem[r];
  }
  __syncthreads();
  if (tid < 16) {
    float o = blin[0];
#pragma unroll 8
    for (int h = 0; h < HDIM; ++h) o += sM32[tid * HDIM + h] * Wlin[h];
    const float e = o - yv[row0 + tid];
    sqerr[row0 + tid] = e * e;
  }
}

// Deterministic single-block tree reduction: mean of n squared errors.
__global__ __launch_bounds__(256)
void loss_reduce_kernel(const float* __restrict__ part, float* __restrict__ out, int n) {
  __shared__ float s[256];
  float v = 0.0f;
  for (int i = threadIdx.x; i < n; i += 256) v += part[i];
  s[threadIdx.x] = v;
  __syncthreads();
  for (int off = 128; off > 0; off >>= 1) {
    if ((int)threadIdx.x < off) s[threadIdx.x] += s[threadIdx.x + off];
    __syncthreads();
  }
  if (threadIdx.x == 0) out[0] = s[0] * (1.0f / (float)n);
}

extern "C" void kernel_launch(void* const* d_in, const int* in_sizes, int n_in,
                              void* d_out, int out_size, void* d_ws, size_t ws_size,
                              hipStream_t stream) {
  const float* x      = (const float*)d_in[0];
  const float* yv     = (const float*)d_in[1];
  const float* h0_mem = (const float*)d_in[2];
  const float* h0_spk = (const float*)d_in[3];
  const float* h0_b   = (const float*)d_in[4];
  const float* W1x    = (const float*)d_in[5];
  const float* b1x    = (const float*)d_in[6];
  const float* WtauM  = (const float*)d_in[7];
  const float* btauM  = (const float*)d_in[8];
  const float* WtauA  = (const float*)d_in[9];
  const float* btauA  = (const float*)d_in[10];
  const float* Wlin   = (const float*)d_in[11];
  const float* blin   = (const float*)d_in[12];

  const int Bsz = in_sizes[1];                   // 2048
  const int S   = in_sizes[0] / (Bsz * DDIM);    // 1024

  float* sqerr = (float*)d_ws;                   // [B] floats of scratch

  dim3 block(256);
  dim3 grid(Bsz / 16);
  srnn_scan_kernel<<<grid, block, 0, stream>>>(x, yv, h0_mem, h0_spk, h0_b,
                                               W1x, b1x, WtauM, btauM, WtauA, btauA,
                                               Wlin, blin, sqerr, S, Bsz);
  loss_reduce_kernel<<<1, 256, 0, stream>>>(sqerr, (float*)d_out, Bsz);
}
